// SharedGCNEncoder_17910013624521
// MI455X (gfx1250) — compile-verified, matrix operand order checked by reading, not run on hardware
//
#include <hip/hip_runtime.h>
#include <hip/hip_bf16.h>
#include <math.h>

// ---------------------------------------------------------------------------
// SharedGCNEncoder forward for MI455X (gfx1250, wave32, WMMA)
//   x = (data + alpha*gauss) * drop(0.5) * drop(0.5)
//   h = x @ W                      -> bf16 WMMA 16x16x32, f32 accum
//   agg = segment_sum(edge_drop(vals) * h[col], row)  -> f32 L2 atomics
//   out = BN(elu(agg))             -> two vectorized streaming passes
//
// Fragment layout (ISA 7.12.2, 16-bit A 16x32):
//   lane = half*16 + m,  slot p = (k&7) + (k>=16 ? 8 : 0),  half=(k>>3)&1
//   => fragments are contiguous 32B per lane (b128-pair loads, no shuffles).
// W pre-swizzled into this layout in global (L1-resident). A is swizzle-staged
// through *wave-private* LDS regions: the per-wave in-order DS guarantee
// replaces all block barriers in the K-loop. B streams through two explicitly
// named register buffers (unroll-by-2) so loads overlap the WMMA chain.
// ---------------------------------------------------------------------------

typedef __bf16 v16bf __attribute__((ext_vector_type(16)));
typedef float  v8f   __attribute__((ext_vector_type(8)));

#define D_IN   512
#define D_H    128
#define NKT    (D_IN / 32)      // 16 K-slabs
#define ALPHA  0.01f
#define KEEP_E 0.6f
#define BN_EPS 1e-5f

__device__ __forceinline__ unsigned pcg_hash(unsigned x) {
    x = x * 747796405u + 2891336453u;
    unsigned w = ((x >> ((x >> 28u) + 4u)) ^ x) * 277803737u;
    return (w >> 22u) ^ w;
}
__device__ __forceinline__ float u01(unsigned h) {
    return (float)(h >> 8) * (1.0f / 16777216.0f);
}
// One hash per element: gaussian via central-limit popcount (binomial(32,1/2)
// ~ N(16, 8)), both p=0.5 dropout draws from the top two bits (product mask:
// keep-both with P=0.25, scale 1/0.25 = 4).
__device__ __forceinline__ float noise_drop(float d, unsigned g) {
    unsigned h = pcg_hash(g);
    float gauss = ((float)__popc(h) - 16.0f) * 0.35355339f;   // 1/sqrt(8)
    float m = ((h >> 30) == 3u) ? 4.0f : 0.0f;
    return (d + ALPHA * gauss) * m;
}

// ---------------------------------------------------------------------------
// Kernel 0: one-time swizzle of W [512,128] f32 -> bf16 fragment table:
//   Wsw[((kt*8 + ct)*32 + lane)*16 + p]
// ---------------------------------------------------------------------------
__global__ __launch_bounds__(256)
void swizzle_W(const float* __restrict__ W, __bf16* __restrict__ Wsw)
{
    int idx = blockIdx.x * blockDim.x + threadIdx.x;   // 0..65535
    int k   = idx >> 7;          // 0..511
    int col = idx & 127;
    int kt  = k >> 5, kin = k & 31;
    int ct  = col >> 4, n = col & 15;
    int half = (kin >> 3) & 1;
    int p    = (kin & 7) + ((kin >= 16) ? 8 : 0);
    int lane = half * 16 + n;
    Wsw[((size_t)(kt * 8 + ct) * 32 + lane) * 16 + p] = (__bf16)W[(size_t)k * D_H + col];
}

// ---------------------------------------------------------------------------
// Kernel 1: noise + double dropout + GEMM via WMMA.
// Block = 256 threads = 8 waves; each wave independently owns a 32-row tile
// (2 A fragments, 16 accumulators) and stages its own rows into its private
// LDS region -- no __syncthreads in the K-loop.
// ---------------------------------------------------------------------------
__global__ __launch_bounds__(256)
void gemm_noise_wmma(const float* __restrict__ data, const __bf16* __restrict__ Wsw,
                     float* __restrict__ h, int N)
{
    __shared__ __align__(32) __bf16 sA[16][32][16];  // [row tile][lane][slot], 16KB

    const int tid  = threadIdx.x;
    const int wave = tid >> 5;       // wave32
    const int lane = tid & 31;
    const int half = lane >> 4;
    const int l16  = lane & 15;
    const int rowB = blockIdx.x * 256;
    const int rowW = rowB + wave * 32;           // this wave's 32-row tile
    const bool full = (rowW + 32) <= N;

    // Fixed per-lane staging coordinates: 8 float4s = 32 elements of this
    // wave's 32x32 A slab. (r,k) -> swizzled LDS slot is kt-invariant.
    // i4 = i*32 + lane: 8 lanes cover one row's 128B line (coalesced).
    int srow[8], skk[8];
    __bf16* sdst[8];
    #pragma unroll
    for (int i = 0; i < 8; ++i) {
        int i4 = i * 32 + lane;        // 0..255 float4 tiles of the wave slab
        int r  = i4 >> 3;              // 0..31 local row
        int k  = (i4 & 7) * 4;         // 0,4,...,28
        srow[i] = r; skk[i] = k;
        int hf = (k >> 3) & 1;
        int p  = (k & 7) + ((k >= 16) ? 8 : 0);
        sdst[i] = &sA[wave * 2 + (r >> 4)][hf * 16 + (r & 15)][p];
    }

    uint2 areg[8];
    auto stage_load = [&](int kt) {
        #pragma unroll
        for (int i = 0; i < 8; ++i) {
            int grow = rowW + srow[i];
            float4 d = make_float4(0.f, 0.f, 0.f, 0.f);
            if (full || grow < N)
                d = *(const float4*)(data + (size_t)grow * D_IN + kt * 32 + skk[i]);
            float xs[4] = {d.x, d.y, d.z, d.w};
            union { __bf16 b[4]; uint2 u; } pk;
            unsigned gbase = (unsigned)grow * (unsigned)D_IN + (unsigned)(kt * 32 + skk[i]);
            #pragma unroll
            for (int j = 0; j < 4; ++j)
                pk.b[j] = (__bf16)noise_drop(xs[j], gbase + (unsigned)j);
            areg[i] = pk.u;
        }
    };

    v8f c[2][8] = {};
    stage_load(0);

    for (int kt = 0; kt < NKT; ++kt) {
        #pragma unroll
        for (int i = 0; i < 8; ++i)
            *(uint2*)sdst[i] = areg[i];          // ds_store_b64 into own region

        if (kt + 1 < NKT) stage_load(kt + 1);    // overlap next slab with WMMA

        // A fragments: per-wave DS in-order guarantee covers store->load.
        v16bf a0 = *(const v16bf*)(&sA[wave * 2 + 0][lane][0]);   // 2x ds_load_b128
        v16bf a1 = *(const v16bf*)(&sA[wave * 2 + 1][lane][0]);

        // B fragments: two named buffers, unroll-by-2 -> while WMMAs consume
        // one buffer the other buffer's loads are in flight (partial waits).
        const __bf16* wp = Wsw + (size_t)kt * 8 * 32 * 16 + (size_t)lane * 16;
        v16bf b0 = *(const v16bf*)(wp);                           // ct = 0
        v16bf b1 = *(const v16bf*)(wp + (size_t)1 * 32 * 16);     // ct = 1
        #pragma unroll
        for (int ct = 0; ct < 8; ct += 2) {
            c[0][ct] = __builtin_amdgcn_wmma_f32_16x16x32_bf16(
                           false, a0, false, b0, (short)0, c[0][ct], false, false);
            c[1][ct] = __builtin_amdgcn_wmma_f32_16x16x32_bf16(
                           false, a1, false, b0, (short)0, c[1][ct], false, false);
            if (ct + 2 < 8)
                b0 = *(const v16bf*)(wp + (size_t)(ct + 2) * 32 * 16);
            c[0][ct + 1] = __builtin_amdgcn_wmma_f32_16x16x32_bf16(
                               false, a0, false, b1, (short)0, c[0][ct + 1], false, false);
            c[1][ct + 1] = __builtin_amdgcn_wmma_f32_16x16x32_bf16(
                               false, a1, false, b1, (short)0, c[1][ct + 1], false, false);
            if (ct + 3 < 8)
                b1 = *(const v16bf*)(wp + (size_t)(ct + 3) * 32 * 16);
        }
    }

    // Store C: VGPR r -> M = half*8 + r, lane&15 -> N
    #pragma unroll
    for (int t = 0; t < 2; ++t) {
        #pragma unroll
        for (int ct = 0; ct < 8; ++ct) {
            #pragma unroll
            for (int r = 0; r < 8; ++r) {
                int grow = rowW + t * 16 + half * 8 + r;
                if (full || grow < N)
                    h[(size_t)grow * D_H + ct * 16 + l16] = c[t][ct][r];
            }
        }
    }
}

// ---------------------------------------------------------------------------
// Kernel 2: edge-parallel SpMM with edge dropout. One wave per edge; each
// lane moves 4 features (float4 gather from L2-resident h, 4 native f32
// atomics into agg).
// ---------------------------------------------------------------------------
__global__ __launch_bounds__(256)
void spmm_edges(const long long* __restrict__ idx, const float* __restrict__ vals,
                const float* __restrict__ h, float* __restrict__ agg, int E)
{
    long long t = (long long)blockIdx.x * blockDim.x + threadIdx.x;
    int e = (int)(t >> 5);
    if (e >= E) return;
    int lane = (int)(t & 31);

    unsigned he = pcg_hash((unsigned)e ^ 0xA511E9B3u);
    if (u01(he) >= KEEP_E) return;
    float v = vals[e] * (1.0f / KEEP_E);

    long long row = idx[e];
    long long col = idx[(long long)E + e];

    float4 hv = *((const float4*)(h + (size_t)col * D_H) + lane);
    float* dst = agg + (size_t)row * D_H + lane * 4;
    __hip_atomic_fetch_add(dst + 0, v * hv.x, __ATOMIC_RELAXED, __HIP_MEMORY_SCOPE_AGENT);
    __hip_atomic_fetch_add(dst + 1, v * hv.y, __ATOMIC_RELAXED, __HIP_MEMORY_SCOPE_AGENT);
    __hip_atomic_fetch_add(dst + 2, v * hv.z, __ATOMIC_RELAXED, __HIP_MEMORY_SCOPE_AGENT);
    __hip_atomic_fetch_add(dst + 3, v * hv.w, __ATOMIC_RELAXED, __HIP_MEMORY_SCOPE_AGENT);
}

// ---------------------------------------------------------------------------
// Kernel 3: ELU + per-channel sum/sumsq, float4-vectorized. Grid stride in
// float4 units is a multiple of 32, so each thread owns a fixed group of 4
// consecutive channels; block-local reduce, one atomic per channel per block.
// ---------------------------------------------------------------------------
__global__ __launch_bounds__(256)
void elu_stats(const float4* __restrict__ agg4, float4* __restrict__ feat4,
               float* __restrict__ stats, long long total4)
{
    __shared__ float sS[256][4];
    __shared__ float sQ[256][4];
    float s[4] = {0.f, 0.f, 0.f, 0.f}, q[4] = {0.f, 0.f, 0.f, 0.f};
    for (long long i = (long long)blockIdx.x * blockDim.x + threadIdx.x;
         i < total4; i += (long long)gridDim.x * blockDim.x) {
        float4 x = agg4[i];
        float4 f;
        f.x = (x.x > 0.f) ? x.x : (__expf(x.x) - 1.f);
        f.y = (x.y > 0.f) ? x.y : (__expf(x.y) - 1.f);
        f.z = (x.z > 0.f) ? x.z : (__expf(x.z) - 1.f);
        f.w = (x.w > 0.f) ? x.w : (__expf(x.w) - 1.f);
        feat4[i] = f;
        s[0] += f.x; q[0] += f.x * f.x;
        s[1] += f.y; q[1] += f.y * f.y;
        s[2] += f.z; q[2] += f.z * f.z;
        s[3] += f.w; q[3] += f.w * f.w;
    }
    #pragma unroll
    for (int j = 0; j < 4; ++j) { sS[threadIdx.x][j] = s[j]; sQ[threadIdx.x][j] = q[j]; }
    __syncthreads();
    if (threadIdx.x < 128) {
        int c = threadIdx.x, grp = c >> 2, comp = c & 3;
        float ss = 0.f, qq = 0.f;
        #pragma unroll
        for (int j = 0; j < 8; ++j) {
            ss += sS[grp + 32 * j][comp];
            qq += sQ[grp + 32 * j][comp];
        }
        __hip_atomic_fetch_add(&stats[c],       ss, __ATOMIC_RELAXED, __HIP_MEMORY_SCOPE_AGENT);
        __hip_atomic_fetch_add(&stats[128 + c], qq, __ATOMIC_RELAXED, __HIP_MEMORY_SCOPE_AGENT);
    }
}

// Kernel 4: fold stats into scale/shift (128 threads).
__global__ void bn_finalize(float* __restrict__ stats, const float* __restrict__ gamma,
                            const float* __restrict__ beta, float invN)
{
    int c = threadIdx.x;
    if (c < 128) {
        float mean = stats[c] * invN;
        float var  = stats[128 + c] * invN - mean * mean;
        float scale = gamma[c] * __frsqrt_rn(var + BN_EPS);
        stats[256 + c] = scale;
        stats[384 + c] = beta[c] - mean * scale;
    }
}

// Kernel 5: apply affine BN in place, float4-vectorized.
__global__ __launch_bounds__(256)
void bn_apply(float4* __restrict__ feat4, const float* __restrict__ stats, long long total4)
{
    const float4* scale4 = (const float4*)(stats + 256);
    const float4* shift4 = (const float4*)(stats + 384);
    for (long long i = (long long)blockIdx.x * blockDim.x + threadIdx.x;
         i < total4; i += (long long)gridDim.x * blockDim.x) {
        int g = (int)(i & 31);
        float4 f  = feat4[i];
        float4 sc = scale4[g];
        float4 sh = shift4[g];
        f.x = f.x * sc.x + sh.x;
        f.y = f.y * sc.y + sh.y;
        f.z = f.z * sc.z + sh.z;
        f.w = f.w * sc.w + sh.w;
        feat4[i] = f;
    }
}

// ---------------------------------------------------------------------------
extern "C" void kernel_launch(void* const* d_in, const int* in_sizes, int n_in,
                              void* d_out, int out_size, void* d_ws, size_t ws_size,
                              hipStream_t stream)
{
    const float*     data  = (const float*)d_in[0];
    const long long* adj   = (const long long*)d_in[1];  // int64 [2,E]
    const float*     vals  = (const float*)d_in[2];
    const float*     W     = (const float*)d_in[3];
    const float*     gamma = (const float*)d_in[4];
    const float*     beta  = (const float*)d_in[5];

    const int N = in_sizes[0] / D_IN;
    const int E = in_sizes[2];

    float*  h     = (float*)d_ws;                 // [N,128] f32
    float*  agg   = h + (size_t)N * D_H;          // [N,128] f32
    float*  stats = agg + (size_t)N * D_H;        // 512 f32: sum/sumsq/scale/shift
    __bf16* Wsw   = (__bf16*)(stats + 512);       // 512*128 bf16 fragment table

    hipMemsetAsync(agg,   0, (size_t)N * D_H * sizeof(float), stream);
    hipMemsetAsync(stats, 0, 512 * sizeof(float), stream);

    swizzle_W<<<dim3(256), 256, 0, stream>>>(W, Wsw);

    dim3 gGrid((N + 255) / 256);
    gemm_noise_wmma<<<gGrid, 256, 0, stream>>>(data, Wsw, h, N);

    long long spThreads = (long long)E * 32;
    dim3 sGrid((unsigned)((spThreads + 255) / 256));
    spmm_edges<<<sGrid, 256, 0, stream>>>(adj, vals, h, agg, E);

    long long total4 = ((long long)N * D_H) / 4;
    elu_stats<<<1024, 256, 0, stream>>>((const float4*)agg, (float4*)d_out, stats, total4);
    bn_finalize<<<1, 128, 0, stream>>>(stats, gamma, beta, 1.0f / (float)N);
    bn_apply<<<2048, 256, 0, stream>>>((float4*)d_out, stats, total4);
}
